// DilatedAttention_14319420964897
// MI455X (gfx1250) — matrix operand order repaired and checked by reading, not compiled, not run on hardware
//
#include <hip/hip_runtime.h>

// ---------------- problem constants (from reference) ----------------
#define Hh 8
#define Lh 2048
#define Dh 128
#define NBH 32          // B*H
#define BR 128          // q rows per block
#define BCk 64          // k rows per chunk
#define NW 8            // waves per block (256 threads, wave32)

typedef __attribute__((ext_vector_type(16))) __bf16 v16bf;
typedef __attribute__((ext_vector_type(8)))  float  v8f;
typedef __attribute__((ext_vector_type(4)))  __bf16 v4bf;

union Frag16 { v16bf v; uint4 q[2]; };

__device__ __forceinline__ __bf16 f2bf(float f) {
  unsigned u = __builtin_bit_cast(unsigned, f);
  unsigned r = u + 0x7FFFu + ((u >> 16) & 1u);   // round-to-nearest-even
  unsigned short h = (unsigned short)(r >> 16);
  return __builtin_bit_cast(__bf16, h);
}

// A-matrix 16x32 bf16 fragment from a row-major LDS row pointer.
// lane<16: K = {0..7, 16..23}; lane>=16: K = {8..15, 24..31} (ISA 7.12.2).
__device__ __forceinline__ v16bf load_fragA(const __bf16* row, int kchunk, int half) {
  Frag16 f;
  const __bf16* p = row + kchunk * 32 + half * 8;
  f.q[0] = *(const uint4*)p;
  f.q[1] = *(const uint4*)(p + 16);
  return f.v;
}

// B-matrix 32x16 bf16 fragment; column n = lane&15 is a row of the staged tile.
// lane<16: K = {0..7, 16..23}; lane>=16: K = {8..15, 24..31} (matches A layout).
__device__ __forceinline__ v16bf load_fragB(const __bf16* row, int kchunk, int half) {
  Frag16 f;
  const __bf16* p = row + kchunk * 32 + half * 8;
  f.q[0] = *(const uint4*)p;
  f.q[1] = *(const uint4*)(p + 16);
  return f.v;
}

__global__ __launch_bounds__(256)
void dilated_flash_attn(const float* __restrict__ Q,
                        const float* __restrict__ K,
                        const float* __restrict__ V,
                        float* __restrict__ Out,
                        int dr, int accumulate)
{
  __shared__ __align__(16) __bf16 Ks[BCk][136];     // K tile, row-major  (17.4 KB)
  __shared__ __align__(16) __bf16 Vts[Dh][72];      // V tile, transposed (18.4 KB)
  __shared__ __align__(16) __bf16 Ps[NW][16][72];   // per-wave P pad     (18.4 KB)

  const int tid  = threadIdx.x;
  const int wid  = tid >> 5;
  const int lane = tid & 31;
  const int ln   = lane & 15;
  const int half = lane >> 4;

  const int bh = blockIdx.y;
  const int b  = bh >> 3;          // bh / H
  const int j  = bh & 7;           // head
  const int hp = Hh / dr;
  const int r  = j / hp;           // dilation offset for this head
  const int M  = Lh / dr;          // gathered sequence length
  const int q0 = blockIdx.x * BR;  // gathered q-row base for this block

  // ---- Q fragments directly from global (strided rows), fp32 -> bf16 ----
  v16bf qa[4];
  {
    const long grow = (long)(q0 + wid * 16 + ln) * dr + r;
    const float* qp = Q + ((long)bh * Lh + grow) * Dh;
#pragma unroll
    for (int c = 0; c < 4; ++c) {
      const int k0 = c * 32 + half * 8;
      float4 f0 = *(const float4*)(qp + k0);
      float4 f1 = *(const float4*)(qp + k0 + 4);
      float4 f2 = *(const float4*)(qp + k0 + 16);
      float4 f3 = *(const float4*)(qp + k0 + 20);
      v16bf t;
      t[0]  = f2bf(f0.x); t[1]  = f2bf(f0.y); t[2]  = f2bf(f0.z); t[3]  = f2bf(f0.w);
      t[4]  = f2bf(f1.x); t[5]  = f2bf(f1.y); t[6]  = f2bf(f1.z); t[7]  = f2bf(f1.w);
      t[8]  = f2bf(f2.x); t[9]  = f2bf(f2.y); t[10] = f2bf(f2.z); t[11] = f2bf(f2.w);
      t[12] = f2bf(f3.x); t[13] = f2bf(f3.y); t[14] = f2bf(f3.z); t[15] = f2bf(f3.w);
      qa[c] = t;
    }
  }

  v8f o[8];
#pragma unroll
  for (int i = 0; i < 8; ++i) o[i] = (v8f){0.f,0.f,0.f,0.f,0.f,0.f,0.f,0.f};
  float mrow[8], lsum[8];
#pragma unroll
  for (int g = 0; g < 8; ++g) { mrow[g] = -3.0e38f; lsum[g] = 0.f; }

  const int nkc  = M / BCk;
  const int lrow = tid >> 2;       // 0..63 : K/V row within tile
  const int lseg = tid & 3;        // 0..3  : 32-float segment of the row
  const long kvcol = (long)bh * Lh;

  // ---- register double-buffer: prologue loads chunk 0 ----
  float4 kreg[8], vreg[8];
  {
    const long g0 = (long)lrow * dr + r;
    const float4* kp = (const float4*)(K + (kvcol + g0) * Dh + lseg * 32);
    const float4* vp = (const float4*)(V + (kvcol + g0) * Dh + lseg * 32);
#pragma unroll
    for (int i = 0; i < 8; ++i) { kreg[i] = kp[i]; vreg[i] = vp[i]; }
  }

  for (int kc = 0; kc < nkc; ++kc) {
    __syncthreads();               // previous chunk's LDS readers done

    // ---- store staged registers to LDS (fp32 -> bf16) ----
#pragma unroll
    for (int i = 0; i < 8; ++i) {
      float4 kf = kreg[i];
      *(v4bf*)&Ks[lrow][lseg * 32 + i * 4] =
          (v4bf){f2bf(kf.x), f2bf(kf.y), f2bf(kf.z), f2bf(kf.w)};
      float4 vf = vreg[i];
      const int col = lseg * 32 + i * 4;
      Vts[col + 0][lrow] = f2bf(vf.x);
      Vts[col + 1][lrow] = f2bf(vf.y);
      Vts[col + 2][lrow] = f2bf(vf.z);
      Vts[col + 3][lrow] = f2bf(vf.w);
    }

    // ---- issue next chunk's global loads; they complete during compute ----
    if (kc + 1 < nkc) {
      const long ng = (long)((kc + 1) * BCk + lrow) * dr + r;
      const float4* kp = (const float4*)(K + (kvcol + ng) * Dh + lseg * 32);
      const float4* vp = (const float4*)(V + (kvcol + ng) * Dh + lseg * 32);
#pragma unroll
      for (int i = 0; i < 8; ++i) { kreg[i] = kp[i]; vreg[i] = vp[i]; }
    }
    if (kc + 2 < nkc) {            // warm L2 two chunks ahead (global_prefetch_b8)
      const long pg = (long)((kc + 2) * BCk + lrow) * dr + r;
      __builtin_prefetch(K + (kvcol + pg) * Dh + lseg * 32, 0, 1);
      __builtin_prefetch(V + (kvcol + pg) * Dh + lseg * 32, 0, 1);
    }
    __syncthreads();

    // ---- S = Q K^T : software-pipelined B fragments (load n+1 before wmma n) ----
    v8f st[4];
    {
      v16bf bk = load_fragB(&Ks[ln][0], 0, half);
      v8f acc = (v8f){0.f,0.f,0.f,0.f,0.f,0.f,0.f,0.f};
#pragma unroll
      for (int it = 0; it < 16; ++it) {
        const int t = it >> 2, c = it & 3;
        v16bf bk_n = bk;
        if (it < 15) {
          const int nt = (it + 1) >> 2, nc = (it + 1) & 3;
          bk_n = load_fragB(&Ks[nt * 16 + ln][0], nc, half);
        }
        acc = __builtin_amdgcn_wmma_f32_16x16x32_bf16(
                  false, qa[c], false, bk, (short)0, acc, false, false);
        if (c == 3) {
          st[t] = acc;
          acc = (v8f){0.f,0.f,0.f,0.f,0.f,0.f,0.f,0.f};
        }
        bk = bk_n;
      }
    }

    // ---- online softmax (no 1/sqrt(D) scaling, matching reference) ----
    float scale[8], rsum[8];
#pragma unroll
    for (int g = 0; g < 8; ++g) {
      float t0 = fmaxf(fmaxf(st[0][g], st[1][g]), fmaxf(st[2][g], st[3][g]));
      for (int off = 8; off > 0; off >>= 1) t0 = fmaxf(t0, __shfl_xor(t0, off));
      const float mn = fmaxf(mrow[g], t0);
      scale[g] = __expf(mrow[g] - mn);
      mrow[g] = mn;
      rsum[g] = 0.f;
    }
#pragma unroll
    for (int t = 0; t < 4; ++t)
#pragma unroll
      for (int g = 0; g < 8; ++g) {
        const float p = __expf(st[t][g] - mrow[g]);
        rsum[g] += p;
        Ps[wid][g + 8 * half][t * 16 + ln] = f2bf(p);   // C-layout row m = g+8*half
      }
#pragma unroll
    for (int g = 0; g < 8; ++g) {
      float s = rsum[g];
      for (int off = 8; off > 0; off >>= 1) s += __shfl_xor(s, off);
      lsum[g] = lsum[g] * scale[g] + s;
    }
#pragma unroll
    for (int dt = 0; dt < 8; ++dt)
#pragma unroll
      for (int g = 0; g < 8; ++g) o[dt][g] *= scale[g];

    // ---- O += P V : software-pipelined Vt B fragments ----
    const v16bf pa0 = load_fragA(&Ps[wid][ln][0], 0, half);
    const v16bf pa1 = load_fragA(&Ps[wid][ln][0], 1, half);
    {
      v16bf bv = load_fragB(&Vts[ln][0], 0, half);
#pragma unroll
      for (int it = 0; it < 16; ++it) {
        const int dt = it >> 1, c2 = it & 1;
        v16bf bv_n = bv;
        if (it < 15) {
          const int ndt = (it + 1) >> 1, nc2 = (it + 1) & 1;
          bv_n = load_fragB(&Vts[ndt * 16 + ln][0], nc2, half);
        }
        o[dt] = __builtin_amdgcn_wmma_f32_16x16x32_bf16(
                    false, c2 ? pa1 : pa0, false, bv, (short)0, o[dt], false, false);
        bv = bv_n;
      }
    }
  }

  // ---- normalize + scatter to original positions ----
#pragma unroll
  for (int g = 0; g < 8; ++g) {
    const float inv  = 1.0f / lsum[g];
    const long  grow = (long)(q0 + wid * 16 + g + 8 * half) * dr + r;
    float* op = Out + ((long)b * Lh + grow) * (Hh * Dh) + (long)j * Dh + ln;
    if (accumulate) {
#pragma unroll
      for (int dt = 0; dt < 8; ++dt) op[dt * 16] = op[dt * 16] + o[dt][g] * inv;
    } else {
#pragma unroll
      for (int dt = 0; dt < 8; ++dt) op[dt * 16] = o[dt][g] * inv;
    }
  }
}

extern "C" void kernel_launch(void* const* d_in, const int* in_sizes, int n_in,
                              void* d_out, int out_size, void* d_ws, size_t ws_size,
                              hipStream_t stream) {
  (void)in_sizes; (void)n_in; (void)out_size; (void)d_ws; (void)ws_size;
  const float* q = (const float*)d_in[0];
  const float* k = (const float*)d_in[1];
  const float* v = (const float*)d_in[2];
  float* out = (float*)d_out;
  dim3 blk(256);
  // dr=1 writes every output element exactly once -> plain stores (no init needed).
  dilated_flash_attn<<<dim3((Lh / 1) / BR, NBH), blk, 0, stream>>>(q, k, v, out, 1, 0);
  // dr=2/4/8: unique writer per element within each launch; stream order
  // serializes launches -> deterministic read-add-store accumulation.
  dilated_flash_attn<<<dim3((Lh / 2) / BR, NBH), blk, 0, stream>>>(q, k, v, out, 2, 1);
  dilated_flash_attn<<<dim3((Lh / 4) / BR, NBH), blk, 0, stream>>>(q, k, v, out, 4, 1);
  dilated_flash_attn<<<dim3((Lh / 8) / BR, NBH), blk, 0, stream>>>(q, k, v, out, 8, 1);
}
